// ANRTNet_51135880626470
// MI455X (gfx1250) — compile-verified, matrix operand order checked by reading, not run on hardware
//
#include <hip/hip_runtime.h>
#include <hip/hip_bf16.h>

typedef __attribute__((ext_vector_type(16))) __bf16 v16bf;
typedef __attribute__((ext_vector_type(8)))  float  v8f;

struct __align__(16) U128 { unsigned a[4]; };
union Frag { v16bf v; U128 q[2]; };

__device__ __forceinline__ unsigned short f2bf(float f) {
  union { float f; unsigned u; } v; v.f = f;
  unsigned r = v.u + 0x7FFFu + ((v.u >> 16) & 1u);   // round-to-nearest-even
  return (unsigned short)(r >> 16);
}

__device__ __forceinline__ float geluf(float x) {
  return 0.5f * x * (1.0f + erff(x * 0.70710678118654752f));
}

// 16-byte async copy global -> LDS (ASYNCcnt-tracked, CDNA5)
__device__ __forceinline__ void async_cp16(unsigned lds_addr, const void* gptr) {
  asm volatile("global_load_async_to_lds_b128 %0, %1, off"
               :: "v"(lds_addr), "v"(gptr) : "memory");
}
__device__ __forceinline__ void async_wait0() {
  asm volatile("s_wait_asynccnt 0x0" ::: "memory");
}
__device__ __forceinline__ void async_wait4() {
  asm volatile("s_wait_asynccnt 0x4" ::: "memory");
}

// ---------------------------------------------------------------------------
// GEMM: C[M,N] = act(A * B^T + bias + pos + res)
// A: M x K bf16 row-major (lda).  B: N x K bf16 row-major (ldb)  [pre-
// transposed weights / k-matrix / vT].  Outputs: optional f32 C and/or bf16
// Cbf (same ldc).
// Block tile 128x128, 8 waves (4x2), each wave 32x64 = 2x4 WMMA tiles,
// K chunk 32.  Tiles staged with global_load_async_to_lds_b128, double-
// buffered so chunk i+1 copies overlap chunk i WMMAs (in-order ASYNCcnt:
// wait <=4 guarantees the older chunk's 4 copies landed).
// ---------------------------------------------------------------------------
__global__ __launch_bounds__(256) void gemm_bf16_kernel(
    const unsigned short* __restrict__ A, int lda, long sA,
    const unsigned short* __restrict__ B, int ldb, long sB,
    float* C, unsigned short* Cbf, int ldc, long sC,
    const float* __restrict__ bias,
    const float* __restrict__ pos, int posMod,
    const float* res, int ldr, long sR,
    int K, int act)
{
  // 40-short padded rows: 16B-aligned fragment reads, conflict-free banks
  __shared__ unsigned short shA[2][128 * 40];
  __shared__ unsigned short shB[2][128 * 40];
  int z = blockIdx.z;
  A += z * sA; B += z * sB;
  long czoff = (long)z * sC;
  if (res) res += z * sR;
  int row0 = blockIdx.y * 128, col0 = blockIdx.x * 128;
  int t = threadIdx.x, lane = t & 31, wave = t >> 5;
  int wr = (wave & 3) * 32, wc = (wave >> 2) * 64;
  int lr = lane & 15, kh = lane >> 4;
  v8f acc[2][4] = {};

  // staging map: each tile 128x32 -> 2x 16B per thread (rows r and r+64)
  int sr = t >> 2, ss = (t & 3) * 8;
  unsigned lA0[2], lB0[2];
  lA0[0] = (unsigned)(unsigned long long)&shA[0][sr * 40 + ss];
  lA0[1] = (unsigned)(unsigned long long)&shA[1][sr * 40 + ss];
  lB0[0] = (unsigned)(unsigned long long)&shB[0][sr * 40 + ss];
  lB0[1] = (unsigned)(unsigned long long)&shB[1][sr * 40 + ss];
  const unsigned rowHalf = 64 * 40 * 2;  // +64 rows in bytes

  const unsigned short* Ab = A + (long)(row0 + sr) * lda + ss;
  const unsigned short* Bb = B + (long)(col0 + sr) * ldb + ss;
  long a64 = (long)64 * lda, b64 = (long)64 * ldb;

  int nch = K / 32;
  // prologue: chunk 0 -> buffer 0
  async_cp16(lA0[0], Ab);
  async_cp16(lA0[0] + rowHalf, Ab + a64);
  async_cp16(lB0[0], Bb);
  async_cp16(lB0[0] + rowHalf, Bb + b64);

  for (int i = 0; i < nch; i++) {
    if (i + 1 < nch) {           // prefetch chunk i+1 into other buffer
      int nb = (i + 1) & 1;
      int kc = (i + 1) * 32;
      async_cp16(lA0[nb], Ab + kc);
      async_cp16(lA0[nb] + rowHalf, Ab + a64 + kc);
      async_cp16(lB0[nb], Bb + kc);
      async_cp16(lB0[nb] + rowHalf, Bb + b64 + kc);
      async_wait4();             // chunk i resident (in-order completion)
    } else {
      async_wait0();
    }
    __syncthreads();
    int bs = i & 1;
    Frag af[2], bf[4];
#pragma unroll
    for (int mr = 0; mr < 2; mr++) {
      int rr = wr + mr * 16 + lr;
      af[mr].q[0] = *(const U128*)&shA[bs][rr * 40 + kh * 8];
      af[mr].q[1] = *(const U128*)&shA[bs][rr * 40 + 16 + kh * 8];
    }
#pragma unroll
    for (int nc = 0; nc < 4; nc++) {
      int cc = wc + nc * 16 + lr;
      bf[nc].q[0] = *(const U128*)&shB[bs][cc * 40 + kh * 8];
      bf[nc].q[1] = *(const U128*)&shB[bs][cc * 40 + 16 + kh * 8];
    }
#pragma unroll
    for (int mr = 0; mr < 2; mr++)
#pragma unroll
      for (int nc = 0; nc < 4; nc++)
        acc[mr][nc] = __builtin_amdgcn_wmma_f32_16x16x32_bf16(
            false, af[mr].v, false, bf[nc].v, (short)0, acc[mr][nc], false, false);
    __syncthreads();             // buffer bs free for reuse next iteration
  }
#pragma unroll
  for (int mr = 0; mr < 2; mr++)
#pragma unroll
    for (int nc = 0; nc < 4; nc++)
#pragma unroll
      for (int rj = 0; rj < 8; rj++) {
        int row = row0 + wr + mr * 16 + kh * 8 + rj;
        int col = col0 + wc + nc * 16 + lr;
        float v = acc[mr][nc][rj];
        if (bias) v += bias[col];
        if (pos)  v += pos[(row & (posMod - 1)) * ldc + col];   // posMod is 2^k
        if (res)  v += res[(long)row * ldr + col];
        if (act == 1) v = geluf(v);
        long idx = czoff + (long)row * ldc + col;
        if (C)   C[idx] = v;
        if (Cbf) Cbf[idx] = f2bf(v);
      }
}

// ---------------------------------------------------------------------------
// one-time operand conversion: f32 -> bf16 (plain, and transposing KxN->NxK)
// ---------------------------------------------------------------------------
__global__ __launch_bounds__(256) void cvt_kernel(
    const float* __restrict__ in, unsigned short* __restrict__ out, int n)
{
  int id = blockIdx.x * 256 + threadIdx.x;
  if (id < n) out[id] = f2bf(in[id]);
}

__global__ __launch_bounds__(256) void cvtT_kernel(
    const float* __restrict__ in, unsigned short* __restrict__ out, int K, int N)
{
  int id = blockIdx.x * 256 + threadIdx.x;  // id = n*K + k
  if (id < K * N) {
    int nn = id / K, kk = id - nn * K;
    out[id] = f2bf(in[(long)kk * N + nn]);
  }
}

// repack v (cols 512..767 of qkvbf) transposed: vT[b][d][m]
__global__ __launch_bounds__(256) void vrepack_kernel(
    const unsigned short* __restrict__ qkvbf, unsigned short* __restrict__ vT)
{
  int id = blockIdx.x * 256 + threadIdx.x;  // 2*256*1024
  int b = id >> 18, d = (id >> 10) & 255, m = id & 1023;
  vT[id] = qkvbf[((long)(b * 1024 + m)) * 768 + 512 + d];
}

// ---------------------------------------------------------------------------
// Mask MLP: coords(256,2) -> 64 -> 64 (h2 to ws), then mean over coords -> Mf
// ---------------------------------------------------------------------------
__global__ __launch_bounds__(256) void maskmlp1_kernel(
    const float* m0w, const float* m0b, const float* m1w, const float* m1b,
    float* __restrict__ h2out)
{
  int p = threadIdx.x;
  float cx = (float)(p >> 4) * (1.0f / 15.0f);
  float cy = (float)(p & 15) * (1.0f / 15.0f);
  float h1[64];
  for (int j = 0; j < 64; j++)
    h1[j] = fmaxf(cx * m0w[j] + cy * m0w[64 + j] + m0b[j], 0.0f);
  for (int k = 0; k < 64; k++) {
    float a = m1b[k];
    for (int j = 0; j < 64; j++) a += h1[j] * m1w[j * 64 + k];
    h2out[p * 64 + k] = fmaxf(a, 0.0f);
  }
}

__global__ __launch_bounds__(768) void maskmf_kernel(
    const float* __restrict__ h2, const float* m2w, const float* m2b,
    float* __restrict__ Mf)
{
  int jo = threadIdx.x;  // 0..767
  float acc = 0.0f;
  for (int p = 0; p < 256; p++) {
    const float* hp = h2 + p * 64;
    float s = 0.0f;
    for (int k = 0; k < 64; k++) s += hp[k] * m2w[k * 768 + jo];
    acc += s;
  }
  acc = acc * (1.0f / 256.0f) + m2b[jo];
  Mf[jo] = 1.0f / (1.0f + expf(-acc));
}

// ---------------------------------------------------------------------------
// Per-patch DCT / mask / reconstruction / residual (-> bf16) + token energy
// ---------------------------------------------------------------------------
__global__ __launch_bounds__(256) void patch_kernel(
    const float* __restrict__ x, const float* __restrict__ Mf,
    unsigned short* __restrict__ rflat, float* __restrict__ energy)
{
  __shared__ float sD[256], sX[256], sT1[256], sT2[256];
  int blk = blockIdx.x; int b = blk >> 10, n = blk & 1023;
  int gh = n >> 5, gw = n & 31;
  int t = threadIdx.x, i = t >> 4, j = t & 15;
  float dv = cosf(3.14159265358979f * (float)((2 * i + 1) * j) / 32.0f) * 0.353553390593274f;
  if (i == 0) dv *= 0.707106781186548f;
  sD[t] = dv;   // DCT[i][k], k = j
  float sq = 0.0f;
  for (int c = 0; c < 3; c++) {
    sX[t] = x[(((long)(b * 3 + c)) * 512 + gh * 16 + i) * 512 + gw * 16 + j];
    __syncthreads();
    float a = 0.0f;                                   // T1 = X * DCT^T
    for (int k = 0; k < 16; k++) a += sX[i * 16 + k] * sD[j * 16 + k];
    sT1[t] = a; __syncthreads();
    a = 0.0f;                                         // T2 = (DCT * T1) .* Mf
    for (int k = 0; k < 16; k++) a += sD[i * 16 + k] * sT1[k * 16 + j];
    sT2[t] = a * Mf[c * 256 + t]; __syncthreads();
    a = 0.0f;                                         // T3 = T2 * DCT
    for (int k = 0; k < 16; k++) a += sT2[i * 16 + k] * sD[k * 16 + j];
    sT1[t] = a; __syncthreads();
    a = 0.0f;                                         // recon = DCT^T * T3
    for (int k = 0; k < 16; k++) a += sD[k * 16 + i] * sT1[k * 16 + j];
    float rv = sX[t] - a;
    rflat[(long)blk * 768 + c * 256 + t] = f2bf(rv);
    sq += rv * rv;
    __syncthreads();
  }
  sT1[t] = sq; __syncthreads();
  for (int s = 128; s > 0; s >>= 1) { if (t < s) sT1[t] += sT1[t + s]; __syncthreads(); }
  if (t == 0) energy[blk] = sT1[0] * (1.0f / 768.0f);
}

// ---------------------------------------------------------------------------
// LayerNorm (block per token row) -> bf16, optionally fused tau row-dot
// ---------------------------------------------------------------------------
__global__ __launch_bounds__(256) void ln_kernel(
    const float* __restrict__ in, unsigned short* __restrict__ out,
    const float* __restrict__ g, const float* __restrict__ bb,
    const float* __restrict__ tw, const float* __restrict__ tb,
    float* __restrict__ tauOut)
{
  __shared__ float sh[256];
  int r = blockIdx.x, t = threadIdx.x;
  float v = in[(long)r * 256 + t];
  sh[t] = v; __syncthreads();
  for (int s = 128; s > 0; s >>= 1) { if (t < s) sh[t] += sh[t + s]; __syncthreads(); }
  float mean = sh[0] * (1.0f / 256.0f); __syncthreads();
  float d = v - mean;
  sh[t] = d * d; __syncthreads();
  for (int s = 128; s > 0; s >>= 1) { if (t < s) sh[t] += sh[t + s]; __syncthreads(); }
  float var = sh[0] * (1.0f / 256.0f); __syncthreads();
  float hn = d * rsqrtf(var + 1e-5f) * g[t] + bb[t];
  out[(long)r * 256 + t] = f2bf(hn);
  if (tw) {
    sh[t] = hn * tw[t]; __syncthreads();
    for (int s = 128; s > 0; s >>= 1) { if (t < s) sh[t] += sh[t + s]; __syncthreads(); }
    if (t == 0) tauOut[r] = sh[0] + tb[0];
  }
}

// ---------------------------------------------------------------------------
// Gate MLP per token
// ---------------------------------------------------------------------------
__global__ __launch_bounds__(256) void gate_kernel(
    const float* __restrict__ xt, const float* __restrict__ energy,
    const float* g0w, const float* g0b, const float* g1w, const float* g1b,
    float* __restrict__ gates)
{
  int r = blockIdx.x * 256 + threadIdx.x;  // 0..2047
  const float* p = xt + (long)r * 256;
  float s = 0.0f;
  for (int d = 0; d < 256; d++) { float v = p[d]; s += v * v; }
  float nrm = sqrtf(s), e = energy[r];
  float acc = g1b[0];
  for (int j = 0; j < 64; j++) {
    float h = fmaxf(e * g0w[j] + nrm * g0w[64 + j] + g0b[j], 0.0f);
    acc += h * g1w[j];
  }
  gates[r] = 1.0f / (1.0f + expf(-acc));
}

// ---------------------------------------------------------------------------
// Attention weighting -> bf16 W buffer
// ---------------------------------------------------------------------------
__global__ __launch_bounds__(256) void attnw_kernel(
    const float* __restrict__ S, unsigned short* __restrict__ Wb,
    const float* __restrict__ gates, const float* __restrict__ tau)
{
  __shared__ float sh[256];
  int n = blockIdx.x, b = blockIdx.y, t = threadIdx.x;
  const float* row = S + ((long)b * 1024 + n) * 1024;
  unsigned short* wrow = Wb + ((long)b * 1024 + n) * 1024;
  float gn = gates[b * 1024 + n];
  float tv = tau[b * 1024 + n];
  float a[4];
#pragma unroll
  for (int i = 0; i < 4; i++) {
    int m = t + i * 256;
    a[i] = row[m] * 0.0625f * powf(gn * gates[b * 1024 + m], 0.8f);
  }
  float mx = fmaxf(fmaxf(a[0], a[1]), fmaxf(a[2], a[3]));
  sh[t] = mx; __syncthreads();
  for (int s = 128; s > 0; s >>= 1) { if (t < s) sh[t] = fmaxf(sh[t], sh[t + s]); __syncthreads(); }
  mx = sh[0]; __syncthreads();
  float e[4], lsum = 0.0f;
#pragma unroll
  for (int i = 0; i < 4; i++) { e[i] = expf(a[i] - mx); lsum += e[i]; }
  sh[t] = lsum; __syncthreads();
  for (int s = 128; s > 0; s >>= 1) { if (t < s) sh[t] += sh[t + s]; __syncthreads(); }
  float denom = sh[0]; __syncthreads();
  float w[4], wsum = 0.0f;
#pragma unroll
  for (int i = 0; i < 4; i++) {
    float ms = 1.0f / (1.0f + expf(-12.0f * (a[i] - tv)));
    w[i] = ms * e[i] / denom; wsum += w[i];
  }
  sh[t] = wsum; __syncthreads();
  for (int s = 128; s > 0; s >>= 1) { if (t < s) sh[t] += sh[t + s]; __syncthreads(); }
  float tot = sh[0] + 1e-12f;
#pragma unroll
  for (int i = 0; i < 4; i++) wrow[t + i * 256] = f2bf(w[i] / tot);
}

// ---------------------------------------------------------------------------
// Contrast amplification -> bf16 tokens for the conv head
// ---------------------------------------------------------------------------
__global__ __launch_bounds__(256) void amp_kernel(
    const float* __restrict__ xt, unsigned short* __restrict__ xtbf,
    const float* __restrict__ energy)
{
  __shared__ float sh[256];
  __shared__ float sAmp;
  int r = blockIdx.x, t = threadIdx.x;
  float v = xt[(long)r * 256 + t];
  sh[t] = v * v; __syncthreads();
  for (int s = 128; s > 0; s >>= 1) { if (t < s) sh[t] += sh[t + s]; __syncthreads(); }
  if (t == 0) {
    float base = sqrtf(sh[0]);
    int b = r >> 10, n = r & 1023, gh = n >> 5, gw = n & 31;
    const float* eg = energy + b * 1024;
    float e1 = eg[n];
    float p2 = 0.0f; int h2 = gh & ~1, w2 = gw & ~1;
    for (int di = 0; di < 2; di++) for (int dj = 0; dj < 2; dj++)
      p2 += eg[(h2 + di) * 32 + (w2 + dj)];
    p2 *= 0.25f;
    float p4 = 0.0f; int h4 = gh & ~3, w4 = gw & ~3;
    for (int di = 0; di < 4; di++) for (int dj = 0; dj < 4; dj++)
      p4 += eg[(h4 + di) * 32 + (w4 + dj)];
    p4 *= 0.0625f;
    float mean = (e1 + p2 + p4) * (1.0f / 3.0f);
    float c = fabsf(base - mean) / (1e-6f + base + mean);
    sAmp = 1.0f + 2.0f * fmaxf(c - 0.02f, 0.0f);
  }
  __syncthreads();
  xtbf[(long)r * 256 + t] = f2bf(v * sAmp);
}

// ---------------------------------------------------------------------------
// Direct 3x3 conv, (b, n=h*32+w, c) layout, pad 1, optional gelu
// ---------------------------------------------------------------------------
__global__ __launch_bounds__(256) void conv3x3_kernel(
    const float* __restrict__ in, const float* __restrict__ wt,
    const float* __restrict__ bs, float* __restrict__ out, int dogelu)
{
  int id = blockIdx.x * 256 + threadIdx.x;  // 2*1024*128
  int co = id & 127;
  int n  = (id >> 7) & 1023;
  int b  = id >> 17;
  int h = n >> 5, w = n & 31;
  float acc = bs[co];
  for (int ky = 0; ky < 3; ky++) {
    int yy = h + ky - 1; if (yy < 0 || yy > 31) continue;
    for (int kx = 0; kx < 3; kx++) {
      int xx = w + kx - 1; if (xx < 0 || xx > 31) continue;
      const float* ip = in + ((long)(b * 1024 + yy * 32 + xx)) * 128;
      const float* wp = wt + (long)co * 1152 + ky * 3 + kx;
#pragma unroll 8
      for (int ci = 0; ci < 128; ci++) acc += ip[ci] * wp[ci * 9];
    }
  }
  out[id] = dogelu ? geluf(acc) : acc;
}

// ---------------------------------------------------------------------------
// Fused bilinear 16x upsample + ev(sigmoid)/al(softmax) 1x1 convs + combine
// ---------------------------------------------------------------------------
__global__ __launch_bounds__(256) void final_kernel(
    const float* __restrict__ hd,
    const float* evw, const float* evb, const float* alw, const float* alb,
    float* __restrict__ outp)
{
  int id = blockIdx.x * 256 + threadIdx.x;  // 2*512*512
  int b = id >> 18;
  int rem = id & 262143;
  int y = rem >> 9, x = rem & 511;
  float sy = (y + 0.5f) * 0.0625f - 0.5f;
  float sx = (x + 0.5f) * 0.0625f - 0.5f;
  float fy = floorf(sy), fx = floorf(sx);
  float wy = sy - fy, wx = sx - fx;
  int y0 = (int)fy, x0 = (int)fx;
  int y0c = min(max(y0, 0), 31), y1c = min(max(y0 + 1, 0), 31);
  int x0c = min(max(x0, 0), 31), x1c = min(max(x0 + 1, 0), 31);
  const float* p00 = hd + ((long)(b * 1024 + y0c * 32 + x0c)) * 128;
  const float* p01 = hd + ((long)(b * 1024 + y0c * 32 + x1c)) * 128;
  const float* p10 = hd + ((long)(b * 1024 + y1c * 32 + x0c)) * 128;
  const float* p11 = hd + ((long)(b * 1024 + y1c * 32 + x1c)) * 128;
  float w00 = (1.0f - wy) * (1.0f - wx), w01 = (1.0f - wy) * wx;
  float w10 = wy * (1.0f - wx), w11 = wy * wx;
  float ev[4] = {0, 0, 0, 0}, al[4] = {0, 0, 0, 0};
  for (int c = 0; c < 128; c++) {
    float v = w00 * p00[c] + w01 * p01[c] + w10 * p10[c] + w11 * p11[c];
#pragma unroll
    for (int e2 = 0; e2 < 4; e2++) {
      ev[e2] += evw[e2 * 128 + c] * v;
      al[e2] += alw[e2 * 128 + c] * v;
    }
  }
  float mx = -1e30f;
#pragma unroll
  for (int e2 = 0; e2 < 4; e2++) {
    ev[e2] = 1.0f / (1.0f + expf(-(ev[e2] + evb[e2])));
    al[e2] += alb[e2];
    mx = fmaxf(mx, al[e2]);
  }
  float se = 0.0f, ee[4];
#pragma unroll
  for (int e2 = 0; e2 < 4; e2++) { ee[e2] = expf(al[e2] - mx); se += ee[e2]; }
  float m = 0.0f;
#pragma unroll
  for (int e2 = 0; e2 < 4; e2++) m += (ee[e2] / se) * ev[e2];
  outp[id] = m;
}

// ---------------------------------------------------------------------------
extern "C" void kernel_launch(void* const* d_in, const int* in_sizes, int n_in,
                              void* d_out, int out_size, void* d_ws, size_t ws_size,
                              hipStream_t stream)
{
  (void)in_sizes; (void)n_in; (void)out_size; (void)ws_size;
  int ix = 0;
  auto nf = [&]() { return (const float*)d_in[ix++]; };
  const float* x   = nf();
  const float* m0w = nf(); const float* m0b = nf();
  const float* m1w = nf(); const float* m1b = nf();
  const float* m2w = nf(); const float* m2b = nf();
  const float* emw = nf(); const float* emb = nf();
  const float* pos = nf();
  const float* g0w = nf(); const float* g0b = nf();
  const float* g1w = nf(); const float* g1b = nf();
  const float* L[6][14];
  for (int l = 0; l < 6; l++) for (int k = 0; k < 14; k++) L[l][k] = nf();
  const float* c1w = nf(); const float* c1b = nf();
  const float* c2w = nf(); const float* c2b = nf();
  const float* c3w = nf(); const float* c3b = nf();
  const float* evw = nf(); const float* evb = nf();
  const float* alw = nf(); const float* alb = nf();
  const float* mhw = nf(); const float* mhb = nf(); (void)mhw; (void)mhb; // unused by forward

  // byte-based ws allocator, 256B-aligned blocks
  char* base = (char*)d_ws;
  size_t off = 0;
  auto allocB = [&](size_t bytes) { void* p = base + off; off = (off + bytes + 255) & ~(size_t)255; return p; };
  typedef unsigned short us;
  float* Mf     = (float*)allocB(768 * 4);
  float* h2m    = (float*)allocB(256 * 64 * 4);
  us*    rflat  = (us*)allocB((size_t)2048 * 768 * 2);
  float* energy = (float*)allocB(2048 * 4);
  float* xt     = (float*)allocB((size_t)2048 * 256 * 4);
  us*    hnbf   = (us*)allocB((size_t)2048 * 256 * 2);
  us*    qkvbf  = (us*)allocB((size_t)2048 * 768 * 2);
  us*    vT     = (us*)allocB((size_t)2 * 256 * 1024 * 2);
  float* tauB   = (float*)allocB(2048 * 4);
  float* gates  = (float*)allocB(2048 * 4);
  float* S      = (float*)allocB((size_t)2 * 1024 * 1024 * 4);
  us*    Wbf    = (us*)allocB((size_t)2 * 1024 * 1024 * 2);
  us*    attnbf = (us*)allocB((size_t)2048 * 256 * 2);
  us*    ffnhbf = (us*)allocB((size_t)2048 * 1024 * 2);
  us*    xtbf   = (us*)allocB((size_t)2048 * 256 * 2);
  float* hd1    = (float*)allocB((size_t)2048 * 128 * 4);
  float* hd2    = (float*)allocB((size_t)2048 * 128 * 4);
  float* hd3    = (float*)allocB((size_t)2048 * 128 * 4);
  us*    embT   = (us*)allocB((size_t)256 * 768 * 2);
  us*    qkvwT  = (us*)allocB((size_t)768 * 256 * 2);
  us*    projwT = (us*)allocB((size_t)256 * 256 * 2);
  us*    ffn1wT = (us*)allocB((size_t)1024 * 256 * 2);
  us*    ffn2wT = (us*)allocB((size_t)256 * 1024 * 2);
  us*    c1wbf  = (us*)allocB((size_t)128 * 256 * 2);

  auto gemm = [&](const us* Ap, int lda, long sAa,
                  const us* Bp, int ldb, long sBb,
                  float* Cp, us* Cbf, int ldc, long sCc,
                  const float* bias, const float* posp, int posMod,
                  const float* resp, int ldr, long sRr,
                  int M, int N, int K, int act, int batch) {
    dim3 g(N / 128, M / 128, batch);
    gemm_bf16_kernel<<<g, 256, 0, stream>>>(Ap, lda, sAa, Bp, ldb, sBb,
                                            Cp, Cbf, ldc, sCc, bias, posp, posMod,
                                            resp, ldr, sRr, K, act);
  };
  auto cvtT = [&](const float* in, us* out, int K, int N) {
    cvtT_kernel<<<(K * N + 255) / 256, 256, 0, stream>>>(in, out, K, N);
  };

  // one-time weight conversion (bf16, transposed to NxK)
  cvtT(emw, embT, 768, 256);
  cvt_kernel<<<(128 * 256 + 255) / 256, 256, 0, stream>>>(c1w, c1wbf, 128 * 256);

  // front-end
  maskmlp1_kernel<<<1, 256, 0, stream>>>(m0w, m0b, m1w, m1b, h2m);
  maskmf_kernel<<<1, 768, 0, stream>>>(h2m, m2w, m2b, Mf);
  patch_kernel<<<2048, 256, 0, stream>>>(x, Mf, rflat, energy);
  // xt = rflat @ embed.w + embed.b + pos   (f32 token stream)
  gemm(rflat, 768, 0, embT, 768, 0, xt, nullptr, 256, 0, emb, pos, 1024,
       nullptr, 0, 0, 2048, 256, 768, 0, 1);
  gate_kernel<<<8, 256, 0, stream>>>(xt, energy, g0w, g0b, g1w, g1b, gates);

  // transformer layers
  for (int l = 0; l < 6; l++) {
    cvtT(L[l][2], qkvwT, 256, 768);
    cvtT(L[l][4], projwT, 256, 256);
    cvtT(L[l][8], ffn1wT, 256, 1024);
    cvtT(L[l][10], ffn2wT, 1024, 256);
    ln_kernel<<<2048, 256, 0, stream>>>(xt, hnbf, L[l][0], L[l][1], L[l][12], L[l][13], tauB);
    // qkv (bf16 out only)
    gemm(hnbf, 256, 0, qkvwT, 256, 0, nullptr, qkvbf, 768, 0, L[l][3], nullptr, 1,
         nullptr, 0, 0, 2048, 768, 256, 0, 1);
    vrepack_kernel<<<2048, 256, 0, stream>>>(qkvbf, vT);
    // S = q @ k^T  (per batch, f32)
    gemm(qkvbf, 768, (long)1024 * 768, qkvbf + 256, 768, (long)1024 * 768,
         S, nullptr, 1024, (long)1024 * 1024, nullptr, nullptr, 1,
         nullptr, 0, 0, 1024, 1024, 256, 0, 2);
    attnw_kernel<<<dim3(1024, 2), 256, 0, stream>>>(S, Wbf, gates, tauB);
    // attn = W @ v   (bf16 out)
    gemm(Wbf, 1024, (long)1024 * 1024, vT, 1024, (long)256 * 1024,
         nullptr, attnbf, 256, (long)1024 * 256, nullptr, nullptr, 1,
         nullptr, 0, 0, 1024, 256, 1024, 0, 2);
    // xt += attn @ proj.w + proj.b
    gemm(attnbf, 256, 0, projwT, 256, 0, xt, nullptr, 256, 0, L[l][5], nullptr, 1,
         xt, 256, 0, 2048, 256, 256, 0, 1);
    ln_kernel<<<2048, 256, 0, stream>>>(xt, hnbf, L[l][6], L[l][7], nullptr, nullptr, nullptr);
    gemm(hnbf, 256, 0, ffn1wT, 256, 0, nullptr, ffnhbf, 1024, 0, L[l][9], nullptr, 1,
         nullptr, 0, 0, 2048, 1024, 256, 1, 1);                 // gelu
    gemm(ffnhbf, 1024, 0, ffn2wT, 1024, 0, xt, nullptr, 256, 0, L[l][11], nullptr, 1,
         xt, 256, 0, 2048, 256, 1024, 0, 1);
  }

  // head
  amp_kernel<<<2048, 256, 0, stream>>>(xt, xtbf, energy);
  gemm(xtbf, 256, 0, c1wbf, 256, 0, hd1, nullptr, 128, 0, c1b, nullptr, 1,
       nullptr, 0, 0, 2048, 128, 256, 1, 1);                    // 1x1 conv + gelu
  conv3x3_kernel<<<1024, 256, 0, stream>>>(hd1, c2w, c2b, hd2, 1);
  conv3x3_kernel<<<1024, 256, 0, stream>>>(hd2, c3w, c3b, hd3, 0);
  final_kernel<<<2048, 256, 0, stream>>>(hd3, evw, evb, alw, alb, (float*)d_out);
}
